// AttentionBlock_63797444215711
// MI455X (gfx1250) — matrix operand order repaired and checked by reading, not compiled, not run on hardware
//
#include <hip/hip_runtime.h>

typedef __attribute__((ext_vector_type(16))) _Float16 v16h;
typedef __attribute__((ext_vector_type(8)))  _Float16 v8h;
typedef __attribute__((ext_vector_type(8)))  float    v8f;

#define B_SZ   8
#define C_CH   512
#define N_TOK  4096
#define GROUPS 8

__device__ __forceinline__ int lane_id() { return (int)(threadIdx.x & 31); }

// A-fragment (16x32 f16, M x K): lane l<16 -> M=l, K=0..15; lane l>=16 -> K=16..31
// src is row-major [m][k], pitch in halves, k contiguous.
__device__ __forceinline__ v16h load_a_frag(const _Float16* base, int pitch, int kbase) {
  int l = lane_id(); int m = l & 15; int g = l >> 4;
  const _Float16* p = base + (size_t)m * pitch + kbase + 16 * g;
  v8h lo = *(const v8h*)(p);
  v8h hi = *(const v8h*)(p + 8);
  v16h r;
#pragma unroll
  for (int i = 0; i < 8; ++i) { r[i] = lo[i]; r[i + 8] = hi[i]; }
  return r;
}

// B-fragment (32x16 f16, K x N): lane l -> column n=l&15; lanes 0-15 hold K=0..15, 16-31 hold K=16..31.
// src stored as Bt[n][k] row-major (k contiguous per column), pitch in halves.
__device__ __forceinline__ v16h load_b_frag(const _Float16* base, int pitch, int kbase) {
  int l = lane_id(); int n = l & 15; int g = l >> 4;
  return *(const v16h*)(base + (size_t)n * pitch + kbase + 16 * g);
}

__device__ __forceinline__ v8f wmma_f16(v16h a, v16h b, v8f c) {
  return __builtin_amdgcn_wmma_f32_16x16x32_f16(false, a, false, b, (short)0, c, false, false);
}

// ---------------- weight f32 -> f16 ----------------
__global__ void cvt_weights(const float* __restrict__ wq, const float* __restrict__ wp,
                            _Float16* __restrict__ wq16, _Float16* __restrict__ wp16) {
  int i = blockIdx.x * 256 + threadIdx.x;
  if (i < 3 * C_CH * C_CH) wq16[i] = (_Float16)wq[i];
  if (i < C_CH * C_CH)     wp16[i] = (_Float16)wp[i];
}

// ---------------- GroupNorm stats: one block per (b, g) ----------------
__global__ void __launch_bounds__(256) gn_stats(const float* __restrict__ x, float* __restrict__ stats) {
  int b = blockIdx.x >> 3, g = blockIdx.x & 7;
  const float* base = x + ((size_t)b * C_CH + g * (C_CH / GROUPS)) * N_TOK;
  float s = 0.f, ss = 0.f;
  for (int idx = threadIdx.x; idx < (C_CH / GROUPS) * N_TOK; idx += 256) {
    float v = base[idx];
    s += v; ss += v * v;
  }
  __shared__ float r1[256], r2[256];
  r1[threadIdx.x] = s; r2[threadIdx.x] = ss;
  __syncthreads();
  for (int off = 128; off > 0; off >>= 1) {
    if ((int)threadIdx.x < off) {
      r1[threadIdx.x] += r1[threadIdx.x + off];
      r2[threadIdx.x] += r2[threadIdx.x + off];
    }
    __syncthreads();
  }
  if (threadIdx.x == 0) {
    const float inv = 1.f / ((C_CH / GROUPS) * (float)N_TOK);
    float mean = r1[0] * inv;
    float var  = r2[0] * inv - mean * mean;
    stats[2 * blockIdx.x]     = mean;
    stats[2 * blockIdx.x + 1] = rsqrtf(var + 1e-5f);
  }
}

// ---------------- GroupNorm apply + transpose: x[b,c,n] -> hT[b,n,c] f16 ----------------
__global__ void __launch_bounds__(256) gn_apply(const float* __restrict__ x,
                                                const float* __restrict__ gamma,
                                                const float* __restrict__ beta,
                                                const float* __restrict__ stats,
                                                _Float16* __restrict__ hT) {
  int b = blockIdx.y;
  int n0 = blockIdx.x * 64;
  __shared__ _Float16 tile[64][72];
  int tr = threadIdx.x >> 6;  // 0..3
  int tc = threadIdx.x & 63;  // 0..63
  for (int cb = 0; cb < 8; ++cb) {      // each 64-channel chunk == one group
    float mean = stats[2 * (b * GROUPS + cb)];
    float rs   = stats[2 * (b * GROUPS + cb) + 1];
#pragma unroll
    for (int ci = 0; ci < 16; ++ci) {
      int cl = ci * 4 + tr;
      int c  = cb * 64 + cl;
      float v = x[((size_t)b * C_CH + c) * N_TOK + n0 + tc];
      v = (v - mean) * rs * gamma[c] + beta[c];
      tile[tc][cl] = (_Float16)v;
    }
    __syncthreads();
#pragma unroll
    for (int ni = 0; ni < 16; ++ni) {
      int nl = ni * 4 + tr;
      hT[((size_t)b * N_TOK + n0 + nl) * C_CH + cb * 64 + tc] = tile[nl][tc];
    }
    __syncthreads();
  }
}

// ---------------- QKV GEMM: qkvT[b,n,o] = sum_c hT[n,c]*w[o,c] (+bias) ----------------
// Writes qT (scaled by C^-0.5), kT as [b, n, c]; v as [b, c, n].
__global__ void __launch_bounds__(256) qkv_gemm(const _Float16* __restrict__ hT,
                                                const _Float16* __restrict__ wq16,
                                                const float* __restrict__ bqkv,
                                                _Float16* __restrict__ qT,
                                                _Float16* __restrict__ kT,
                                                _Float16* __restrict__ vmat) {
  int b = blockIdx.z;
  int n_blk = blockIdx.x * 128;
  int o_blk = blockIdx.y * 64;
  int w = threadIdx.x >> 5;
  int n0 = n_blk + (w & 3) * 32;
  int o0 = o_blk + (w >> 2) * 32;
  const _Float16* abase = hT + ((size_t)b * N_TOK + n0) * C_CH;
  const _Float16* bbase = wq16 + (size_t)o0 * C_CH;
  v8f acc[2][2] = {};
  for (int ks = 0; ks < C_CH / 32; ++ks) {
    v16h a0 = load_a_frag(abase,               C_CH, ks * 32);
    v16h a1 = load_a_frag(abase + 16 * C_CH,   C_CH, ks * 32);
    v16h b0 = load_b_frag(bbase,               C_CH, ks * 32);
    v16h b1 = load_b_frag(bbase + 16 * C_CH,   C_CH, ks * 32);
    acc[0][0] = wmma_f16(a0, b0, acc[0][0]);
    acc[0][1] = wmma_f16(a0, b1, acc[0][1]);
    acc[1][0] = wmma_f16(a1, b0, acc[1][0]);
    acc[1][1] = wmma_f16(a1, b1, acc[1][1]);
  }
  int l = lane_id(); int nc = l & 15; int gg = l >> 4;
  int kind = o_blk / C_CH;                        // 0=q, 1=k, 2=v
  const float qscale = 0.04419417382415922f;      // 512^-0.5
#pragma unroll
  for (int ni = 0; ni < 2; ++ni) {
#pragma unroll
    for (int oi = 0; oi < 2; ++oi) {
      int o = o0 + oi * 16 + nc;
      float bias = bqkv[o];
      if (kind == 2) {
        int c = o - 2 * C_CH;
        v8h pk;
#pragma unroll
        for (int r = 0; r < 8; ++r) pk[r] = (_Float16)(acc[ni][oi][r] + bias);
        *(v8h*)(vmat + ((size_t)b * C_CH + c) * N_TOK + n0 + ni * 16 + 8 * gg) = pk;
      } else {
        _Float16* dst = (kind == 0) ? qT : kT;
        float scale = (kind == 0) ? qscale : 1.f;
        int oc = o - kind * C_CH;
#pragma unroll
        for (int r = 0; r < 8; ++r) {
          int tok = n0 + ni * 16 + r + 8 * gg;
          dst[((size_t)b * N_TOK + tok) * C_CH + oc] = (_Float16)((acc[ni][oi][r] + bias) * scale);
        }
      }
    }
  }
}

// ---------------- Fused attention: one block = 32 queries, full softmax rows in LDS ----------------
#define IT_Q   32            // queries per block (2 WMMA row-tiles)
#define SPITCH 4104          // 4096 + 8 halves pad
__global__ void __launch_bounds__(256) attn_fused(const _Float16* __restrict__ qT,
                                                  const _Float16* __restrict__ kT,
                                                  const _Float16* __restrict__ vmat,
                                                  _Float16* __restrict__ attT) {
  __shared__ _Float16 S[IT_Q * SPITCH];   // ~257 KB of the 320 KB WGP LDS
  __shared__ float red[32 * 8];
  __shared__ float rowinv[32];
  int b  = blockIdx.y;
  int i0 = blockIdx.x * IT_Q;
  int w  = threadIdx.x >> 5;
  int l = lane_id(); int nc = l & 15; int gg = l >> 4;

  // Phase 1: scores S[i, j] = sum_c qT[i,c] * kT[j,c]; waves own 512-wide j strips,
  // sweep the two 16-query tiles sequentially so only 16 q A-fragments stay resident.
#pragma unroll 1
  for (int it = 0; it < 2; ++it) {
    const _Float16* qbase = qT + ((size_t)b * N_TOK + i0 + it * 16) * C_CH;
    v16h aq[16];
#pragma unroll
    for (int kk = 0; kk < 16; ++kk) aq[kk] = load_a_frag(qbase, C_CH, kk * 32);
#pragma unroll 1
    for (int jt = 0; jt < 32; ++jt) {
      int j0 = w * 512 + jt * 16;
      const _Float16* kb = kT + ((size_t)b * N_TOK + j0) * C_CH;
      if (jt < 31) {  // prefetch next 16x512 f16 k-tile (global_prefetch_b8 path)
        const char* nx = (const char*)(kb + 16 * C_CH);
#pragma unroll
        for (int pf = 0; pf < 4; ++pf)
          __builtin_prefetch(nx + (size_t)l * 128 + pf * 4096, 0, 1);
      }
      v8f acc = {};
#pragma unroll
      for (int kk = 0; kk < 16; ++kk) {
        v16h bk = load_b_frag(kb, C_CH, kk * 32);
        acc = wmma_f16(aq[kk], bk, acc);
      }
#pragma unroll
      for (int r = 0; r < 8; ++r)
        S[(it * 16 + r + 8 * gg) * SPITCH + j0 + nc] = (_Float16)acc[r];
    }
  }
  __syncthreads();

  // Phase 2: softmax over each of the 32 rows; 8 threads per row, 512 cols each,
  // all LDS traffic vectorized as 128-bit ds ops.
  int row = threadIdx.x >> 3;
  int seg = threadIdx.x & 7;
  _Float16* sr = S + row * SPITCH + seg * 512;
  {
    float m = -1e30f;
#pragma unroll 4
    for (int j = 0; j < 512; j += 8) {
      v8h v = *(const v8h*)(sr + j);
#pragma unroll
      for (int e = 0; e < 8; ++e) m = fmaxf(m, (float)v[e]);
    }
    red[row * 8 + seg] = m;
  }
  __syncthreads();
  float rmax = -1e30f;
#pragma unroll
  for (int s2 = 0; s2 < 8; ++s2) rmax = fmaxf(rmax, red[row * 8 + s2]);
  __syncthreads();
  {
    float sum = 0.f;
#pragma unroll 4
    for (int j = 0; j < 512; j += 8) {
      v8h v = *(const v8h*)(sr + j);
      v8h p;
#pragma unroll
      for (int e = 0; e < 8; ++e) {
        float pe = __expf((float)v[e] - rmax);
        sum += pe;
        p[e] = (_Float16)pe;
      }
      *(v8h*)(sr + j) = p;     // overwrite in place: S now holds P (f16)
    }
    red[row * 8 + seg] = sum;
  }
  __syncthreads();
  if (seg == 0) {
    float s = 0.f;
#pragma unroll
    for (int s2 = 0; s2 < 8; ++s2) s += red[row * 8 + s2];
    rowinv[row] = 1.f / s;
  }
  __syncthreads();

  // Phase 3: outT[i, c] = sum_j P[i,j] * v[c,j]; waves own 64 channels.
  // Each v B-fragment is shared by both 16-query tiles (halves v L2 traffic).
  int c0 = w * 64;
  v8f oacc0[4] = {v8f{}, v8f{}, v8f{}, v8f{}};
  v8f oacc1[4] = {v8f{}, v8f{}, v8f{}, v8f{}};
#pragma unroll 1
  for (int ks = 0; ks < N_TOK / 32; ++ks) {
    v16h ap0 = load_a_frag(S,               SPITCH, ks * 32);
    v16h ap1 = load_a_frag(S + 16 * SPITCH, SPITCH, ks * 32);
#pragma unroll
    for (int ct = 0; ct < 4; ++ct) {
      v16h bv = load_b_frag(vmat + ((size_t)b * C_CH + c0 + ct * 16) * N_TOK, N_TOK, ks * 32);
      oacc0[ct] = wmma_f16(ap0, bv, oacc0[ct]);
      oacc1[ct] = wmma_f16(ap1, bv, oacc1[ct]);
    }
  }
#pragma unroll
  for (int it = 0; it < 2; ++it) {
#pragma unroll
    for (int ct = 0; ct < 4; ++ct) {
#pragma unroll
      for (int r = 0; r < 8; ++r) {
        int i = it * 16 + r + 8 * gg;
        float vres = (it ? oacc1[ct][r] : oacc0[ct][r]) * rowinv[i];
        attT[((size_t)b * N_TOK + i0 + i) * C_CH + c0 + ct * 16 + nc] = (_Float16)vres;
      }
    }
  }
}

// ---------------- Projection + bias + residual, f32 out ----------------
__global__ void __launch_bounds__(256) proj_gemm(const _Float16* __restrict__ attT,
                                                 const _Float16* __restrict__ wp16,
                                                 const float* __restrict__ bproj,
                                                 const float* __restrict__ x,
                                                 float* __restrict__ out) {
  int b = blockIdx.z;
  int n_blk = blockIdx.x * 128;
  int o_blk = blockIdx.y * 64;
  int w = threadIdx.x >> 5;
  int n0 = n_blk + (w & 3) * 32;
  int o0 = o_blk + (w >> 2) * 32;
  const _Float16* abase = wp16 + (size_t)o0 * C_CH;                 // A: rows = o
  const _Float16* bbase = attT + ((size_t)b * N_TOK + n0) * C_CH;  // B: rows = tokens
  v8f acc[2][2] = {};
  for (int ks = 0; ks < C_CH / 32; ++ks) {
    v16h a0 = load_a_frag(abase,             C_CH, ks * 32);
    v16h a1 = load_a_frag(abase + 16 * C_CH, C_CH, ks * 32);
    v16h b0 = load_b_frag(bbase,             C_CH, ks * 32);
    v16h b1 = load_b_frag(bbase + 16 * C_CH, C_CH, ks * 32);
    acc[0][0] = wmma_f16(a0, b0, acc[0][0]);
    acc[0][1] = wmma_f16(a0, b1, acc[0][1]);
    acc[1][0] = wmma_f16(a1, b0, acc[1][0]);
    acc[1][1] = wmma_f16(a1, b1, acc[1][1]);
  }
  int l = lane_id(); int nc = l & 15; int gg = l >> 4;
#pragma unroll
  for (int oi = 0; oi < 2; ++oi) {
#pragma unroll
    for (int ni = 0; ni < 2; ++ni) {
      int tok = n0 + ni * 16 + nc;
#pragma unroll
      for (int r = 0; r < 8; ++r) {
        int o = o0 + oi * 16 + r + 8 * gg;
        size_t idx = ((size_t)b * C_CH + o) * N_TOK + tok;
        out[idx] = acc[oi][ni][r] + bproj[o] + x[idx];
      }
    }
  }
}

extern "C" void kernel_launch(void* const* d_in, const int* in_sizes, int n_in,
                              void* d_out, int out_size, void* d_ws, size_t ws_size,
                              hipStream_t stream) {
  (void)in_sizes; (void)n_in; (void)out_size; (void)ws_size;
  const float* x     = (const float*)d_in[0];
  const float* gamma = (const float*)d_in[1];
  const float* beta  = (const float*)d_in[2];
  const float* wqkv  = (const float*)d_in[3];
  const float* bqkv  = (const float*)d_in[4];
  const float* wproj = (const float*)d_in[5];
  const float* bproj = (const float*)d_in[6];
  float* out = (float*)d_out;

  char* ws = (char*)d_ws;
  size_t off = 0;
  auto walloc = [&](size_t bytes) -> void* {
    off = (off + 255) & ~(size_t)255;
    void* p = ws + off;
    off += bytes;
    return p;
  };
  float*    stats = (float*)walloc(B_SZ * GROUPS * 2 * sizeof(float));
  _Float16* wq16  = (_Float16*)walloc((size_t)3 * C_CH * C_CH * 2);
  _Float16* wp16  = (_Float16*)walloc((size_t)C_CH * C_CH * 2);
  _Float16* hT    = (_Float16*)walloc((size_t)B_SZ * N_TOK * C_CH * 2);
  _Float16* qT    = (_Float16*)walloc((size_t)B_SZ * N_TOK * C_CH * 2);
  _Float16* kT    = (_Float16*)walloc((size_t)B_SZ * N_TOK * C_CH * 2);
  _Float16* vm    = (_Float16*)walloc((size_t)B_SZ * C_CH * N_TOK * 2);
  _Float16* aT    = (_Float16*)walloc((size_t)B_SZ * N_TOK * C_CH * 2);

  cvt_weights<<<(3 * C_CH * C_CH + 255) / 256, 256, 0, stream>>>(wqkv, wproj, wq16, wp16);
  gn_stats<<<B_SZ * GROUPS, 256, 0, stream>>>(x, stats);
  gn_apply<<<dim3(N_TOK / 64, B_SZ), 256, 0, stream>>>(x, gamma, beta, stats, hT);
  qkv_gemm<<<dim3(N_TOK / 128, 3 * C_CH / 64, B_SZ), 256, 0, stream>>>(hT, wq16, bqkv, qT, kT, vm);
  attn_fused<<<dim3(N_TOK / IT_Q, B_SZ), 256, 0, stream>>>(qT, kT, vm, aT);
  proj_gemm<<<dim3(N_TOK / 128, C_CH / 64, B_SZ), 256, 0, stream>>>(aT, wp16, bproj, x, out);
}
